// BertSelfAttention_6073083757125
// MI455X (gfx1250) — compile-verified, matrix operand order checked by reading, not run on hardware
//
#include <hip/hip_runtime.h>
#include <cstdint>
#include <cstddef>

// ---------------------------------------------------------------------------
// MI455X (gfx1250) routed-attention implementation.
// All matmuls use v_wmma_f32_16x16x32_bf16 (wave32 WMMA, fp32 accumulate).
// ---------------------------------------------------------------------------

typedef __attribute__((ext_vector_type(16))) __bf16 v16bf;
typedef __attribute__((ext_vector_type(8)))  __bf16 v8bf;
typedef __attribute__((ext_vector_type(8)))  float  v8f;

constexpr int B_  = 8, S_ = 784, H_ = 1024, NH_ = 16, DH_ = 64;
constexpr int M_  = B_ * S_;            // 6272 tokens
constexpr int MT_ = M_ / 16;            // 392 row tiles
constexpr int NT_ = H_ / 16;            // 64 col tiles
constexpr int KH_ = 595, KPH_ = 608;    // high branch: top 76%, padded to 19*32
constexpr int KL_ = 391, KPL_ = 416;    // low branch: ranks [392,783), padded to 13*32
constexpr int LOWK_ = 392;
constexpr float ROUTE_SCALE_ = 0.102f;
constexpr float SM_SCALE_    = 0.125f;  // 1/sqrt(64)
constexpr float NEG_BIG_     = -3.0e38f;

__device__ __forceinline__ v8f wmma_bf16(v16bf a, v16bf b, v8f c) {
  // D(16x16,f32) = A(16x32,bf16) * B(32x16,bf16) + C
  return __builtin_amdgcn_wmma_f32_16x16x32_bf16(false, a, false, b,
                                                 (short)0, c, false, false);
}

// Per-lane fragment fetch matching the ISA 16-bit A/B layout:
// lane holds K = kb8..kb8+7 and K = 16+kb8..16+kb8+7 of its row/col,
// i.e. two contiguous 16-byte loads at p and p+16 elements.
__device__ __forceinline__ v16bf ld_frag(const __bf16* p) {
  v8bf lo = *(const v8bf*)(p);
  v8bf hi = *(const v8bf*)(p + 16);
  return __builtin_shufflevector(lo, hi, 0,1,2,3,4,5,6,7,8,9,10,11,12,13,14,15);
}

// ---------------------------------------------------------------------------
// 1) fp32 -> bf16 conversion
// ---------------------------------------------------------------------------
__global__ void cvt_bf16_kernel(const float* __restrict__ src,
                                __bf16* __restrict__ dst, int n) {
  int i = blockIdx.x * blockDim.x + threadIdx.x;
  if (i < n) dst[i] = (__bf16)src[i];
}

// ---------------------------------------------------------------------------
// 2) QKV projection: out = hs @ W^T + b, scattered to (B,NH,S,DH) bf16.
//    One 16x16 output tile per wave, K=1024 in 32-wide WMMA steps.
// ---------------------------------------------------------------------------
__global__ __launch_bounds__(256) void qkv_gemm_kernel(
    const __bf16* __restrict__ hsb, const __bf16* __restrict__ wb,
    const float* __restrict__ bq, const float* __restrict__ bk,
    const float* __restrict__ bv,
    __bf16* __restrict__ qb, __bf16* __restrict__ kb, __bf16* __restrict__ vb) {
  int id = blockIdx.x * 8 + (threadIdx.x >> 5);
  if (id >= 3 * MT_ * NT_) return;
  int w   = id / (MT_ * NT_);
  int rem = id - w * (MT_ * NT_);
  int mt  = rem >> 6;          // row tile
  int nt  = rem & 63;          // col tile
  int lane = threadIdx.x & 31;
  int rrow = lane & 15;
  int kb8  = (lane >> 4) << 3;

  const __bf16* Aptr = hsb + (size_t)(mt * 16 + rrow) * H_ + kb8;
  const __bf16* Bptr = wb + (size_t)w * H_ * H_ + (size_t)(nt * 16 + rrow) * H_ + kb8;

  v8f acc = {};
  for (int k0 = 0; k0 < H_; k0 += 32) {
    __builtin_prefetch(Aptr + k0 + 128, 0, 0);   // global_prefetch_b8
    __builtin_prefetch(Bptr + k0 + 128, 0, 0);
    v16bf a = ld_frag(Aptr + k0);
    v16bf b = ld_frag(Bptr + k0);
    acc = wmma_bf16(a, b, acc);
  }

  int n = nt * 16 + (lane & 15);
  const float* bias = (w == 0) ? bq : (w == 1) ? bk : bv;
  float biasv = bias[n];
  __bf16* dst = (w == 0) ? qb : (w == 1) ? kb : vb;
  int head = n >> 6, d = n & 63;
#pragma unroll
  for (int r = 0; r < 8; ++r) {
    int mrow = mt * 16 + r + ((lane >> 4) << 3);
    int bidx = mrow / S_;
    int s    = mrow - bidx * S_;
    dst[(((size_t)(bidx * NH_ + head) * S_ + s) << 6) + d] = (__bf16)(acc[r] + biasv);
  }
}

// ---------------------------------------------------------------------------
// 3) Routing: per (b,h) compute logits row 0, full sort (desc val, asc idx),
//    gather k_sel (row-major), v_sel^T (DH x Kpad), and head-0 mask rows.
// ---------------------------------------------------------------------------
__global__ __launch_bounds__(256) void route_kernel(
    const __bf16* __restrict__ qb, const __bf16* __restrict__ kb,
    const __bf16* __restrict__ vb, const float* __restrict__ amask,
    __bf16* __restrict__ kselh, __bf16* __restrict__ vTh,
    __bf16* __restrict__ ksell, __bf16* __restrict__ vTl,
    float* __restrict__ mh, float* __restrict__ ml) {
  __shared__ float q0[DH_];
  __shared__ float vals[1024];
  __shared__ int   inds[1024];

  int bh = blockIdx.x;                 // 0..127
  int b  = bh >> 4;
  int h  = bh & 15;
  int tid = threadIdx.x;

  if (tid < DH_) q0[tid] = (float)qb[(((size_t)bh * S_) << 6) + tid];
  __syncthreads();

  for (int j = tid; j < 1024; j += 256) {
    if (j < S_) {
      const __bf16* krow = kb + (((size_t)bh * S_ + j) << 6);
      float acc = 0.f;
      for (int d = 0; d < DH_; ++d) acc += q0[d] * (float)krow[d];
      vals[j] = ROUTE_SCALE_ * acc;
    } else {
      vals[j] = NEG_BIG_;
    }
    inds[j] = j;
  }

  // bitonic sort, 1024 elements, descending value / ascending index
  for (int k = 2; k <= 1024; k <<= 1) {
    for (int j = k >> 1; j > 0; j >>= 1) {
      __syncthreads();
      for (int i = tid; i < 1024; i += 256) {
        int ixj = i ^ j;
        if (ixj > i) {
          float va = vals[i], vb2 = vals[ixj];
          int   ia = inds[i], ib = inds[ixj];
          bool a_first = (va > vb2) || (va == vb2 && ia < ib);
          bool up = ((i & k) == 0);
          if (a_first != up) {
            vals[i] = vb2; vals[ixj] = va;
            inds[i] = ib;  inds[ixj] = ia;
          }
        }
      }
    }
  }
  __syncthreads();

  // gather high branch (ranks 0..594), zero-pad to KPH_
  for (int r = tid; r < KPH_; r += 256) {
    __bf16* kd = kselh + (((size_t)bh * KPH_ + r) << 6);
    if (r < KH_) {
      int j = inds[r];
      const __bf16* ks = kb + (((size_t)bh * S_ + j) << 6);
      const __bf16* vs = vb + (((size_t)bh * S_ + j) << 6);
#pragma unroll
      for (int d8 = 0; d8 < 8; ++d8) ((v8bf*)kd)[d8] = ((const v8bf*)ks)[d8];
      for (int d = 0; d < DH_; ++d)
        vTh[((size_t)bh * DH_ + d) * KPH_ + r] = vs[d];
      if (h == 0) mh[b * KPH_ + r] = amask[b * S_ + j];
    } else {
      for (int d = 0; d < DH_; ++d) {
        kd[d] = (__bf16)0.f;
        vTh[((size_t)bh * DH_ + d) * KPH_ + r] = (__bf16)0.f;
      }
      if (h == 0) mh[b * KPH_ + r] = 0.f;
    }
  }
  // gather low branch (ranks 392..782), zero-pad to KPL_
  for (int r = tid; r < KPL_; r += 256) {
    __bf16* kd = ksell + (((size_t)bh * KPL_ + r) << 6);
    if (r < KL_) {
      int j = inds[LOWK_ + r];
      const __bf16* ks = kb + (((size_t)bh * S_ + j) << 6);
      const __bf16* vs = vb + (((size_t)bh * S_ + j) << 6);
#pragma unroll
      for (int d8 = 0; d8 < 8; ++d8) ((v8bf*)kd)[d8] = ((const v8bf*)ks)[d8];
      for (int d = 0; d < DH_; ++d)
        vTl[((size_t)bh * DH_ + d) * KPL_ + r] = vs[d];
      if (h == 0) ml[b * KPL_ + r] = amask[b * S_ + j];
    } else {
      for (int d = 0; d < DH_; ++d) {
        kd[d] = (__bf16)0.f;
        vTl[((size_t)bh * DH_ + d) * KPL_ + r] = (__bf16)0.f;
      }
      if (h == 0) ml[b * KPL_ + r] = 0.f;
    }
  }
}

// ---------------------------------------------------------------------------
// 4/5) Attention branch: one wave handles 16 queries of one (b,h).
//      scores = (q @ k_sel^T)*scale + mask  -> LDS -> softmax -> probs
//      ctx = probs(bf16) @ v_sel  (WMMA with A-fragments from LDS)
// ---------------------------------------------------------------------------
template <int KSEL, int KPAD, bool WRITE_PROBS>
__global__ __launch_bounds__(32) void attn_kernel(
    const __bf16* __restrict__ qb, const __bf16* __restrict__ ksel,
    const __bf16* __restrict__ vT, const float* __restrict__ maskbuf,
    float* __restrict__ probs, float* __restrict__ ctx) {
  extern __shared__ char smem_raw[];
  float*  sc = (float*)smem_raw;                       // 16 x KPAD fp32
  __bf16* pb = (__bf16*)(smem_raw + 16 * KPAD * 4);    // 16 x KPAD bf16

  int blk = blockIdx.x;
  int bh  = blk / (S_ / 16);
  int qt  = blk - bh * (S_ / 16);
  int q0  = qt * 16;
  int b   = bh >> 4;
  int lane = threadIdx.x;
  int rrow = lane & 15;
  int kb8  = (lane >> 4) << 3;
  int hi16 = lane >> 4;

  // q fragments for K = 0..31 and 32..63 (held entire kernel)
  const __bf16* qrow = qb + (((size_t)bh * S_ + q0 + rrow) << 6) + kb8;
  v16bf aq0 = ld_frag(qrow);
  v16bf aq1 = ld_frag(qrow + 32);

  // scores: loop over key tiles
  for (int nt = 0; nt < KPAD / 16; ++nt) {
    const __bf16* krow = ksel + (((size_t)bh * KPAD + nt * 16 + rrow) << 6) + kb8;
    v16bf b0 = ld_frag(krow);
    v16bf b1 = ld_frag(krow + 32);
    v8f acc = {};
    acc = wmma_bf16(aq0, b0, acc);
    acc = wmma_bf16(aq1, b1, acc);
    int n = nt * 16 + rrow;
    float mv = maskbuf[b * KPAD + n];
#pragma unroll
    for (int r = 0; r < 8; ++r) {
      int m = r + (hi16 << 3);
      sc[m * KPAD + n] = acc[r] * SM_SCALE_ + mv;
    }
  }
  __syncthreads();

  // row-wise softmax (pads excluded); write fp32 probs + bf16 copy in LDS
  for (int m = 0; m < 16; ++m) {
    float mx = NEG_BIG_;
    for (int n = lane; n < KSEL; n += 32) mx = fmaxf(mx, sc[m * KPAD + n]);
#pragma unroll
    for (int off = 16; off; off >>= 1) mx = fmaxf(mx, __shfl_xor(mx, off, 32));
    float sum = 0.f;
    for (int n = lane; n < KSEL; n += 32) {
      float e = __expf(sc[m * KPAD + n] - mx);
      sc[m * KPAD + n] = e;
      sum += e;
    }
#pragma unroll
    for (int off = 16; off; off >>= 1) sum += __shfl_xor(sum, off, 32);
    float inv = 1.f / sum;
    for (int n = lane; n < KPAD; n += 32) {
      float p = (n < KSEL) ? sc[m * KPAD + n] * inv : 0.f;
      pb[m * KPAD + n] = (__bf16)p;
      if (WRITE_PROBS && n < KSEL)
        probs[((size_t)bh * S_ + q0 + m) * KSEL + n] = p;
    }
  }
  __syncthreads();

  // ctx = probs @ v_sel : N = DH_ (4 tiles), K = KPAD
  v8f accd[4];
#pragma unroll
  for (int dt = 0; dt < 4; ++dt) accd[dt] = (v8f){};
  for (int kt = 0; kt < KPAD / 32; ++kt) {
    v16bf ap = ld_frag(pb + rrow * KPAD + kt * 32 + kb8);   // ds_load_b128 x2
#pragma unroll
    for (int dt = 0; dt < 4; ++dt) {
      const __bf16* vrow =
          vT + ((size_t)bh * DH_ + dt * 16 + rrow) * KPAD + kt * 32 + kb8;
      v16bf bv = ld_frag(vrow);
      accd[dt] = wmma_bf16(ap, bv, accd[dt]);
    }
  }
#pragma unroll
  for (int dt = 0; dt < 4; ++dt) {
    int d = dt * 16 + rrow;
#pragma unroll
    for (int r = 0; r < 8; ++r) {
      int m = r + (hi16 << 3);
      ctx[(((size_t)bh * S_ + q0 + m) << 6) + d] = accd[dt][r];
    }
  }
}

// ---------------------------------------------------------------------------
// 6) combine: out(b,s,h*64+d) = ctx_hi + 0.2*ctx_lo   (head->channel layout)
// ---------------------------------------------------------------------------
__global__ void combine_kernel(const float* __restrict__ ctxh,
                               const float* __restrict__ ctxl,
                               float* __restrict__ out) {
  int i = blockIdx.x * blockDim.x + threadIdx.x;
  if (i >= M_ * H_) return;
  int c = i & (H_ - 1);
  int t = i >> 10;
  int b = t / S_;
  int s = t - b * S_;
  int h = c >> 6;
  int d = c & 63;
  size_t src = (((size_t)(b * NH_ + h) * S_ + s) << 6) + d;
  out[i] = ctxh[src] + 0.2f * ctxl[src];
}

// ---------------------------------------------------------------------------
extern "C" void kernel_launch(void* const* d_in, const int* in_sizes, int n_in,
                              void* d_out, int out_size, void* d_ws, size_t ws_size,
                              hipStream_t stream) {
  (void)in_sizes; (void)n_in; (void)out_size; (void)ws_size;
  const float* hs    = (const float*)d_in[0];
  const float* amask = (const float*)d_in[1];
  const float* Wq    = (const float*)d_in[2];
  const float* bq    = (const float*)d_in[3];
  const float* Wk    = (const float*)d_in[4];
  const float* bk    = (const float*)d_in[5];
  const float* Wv    = (const float*)d_in[6];
  const float* bv    = (const float*)d_in[7];

  float* out_ctx   = (float*)d_out;
  float* out_probs = out_ctx + (size_t)M_ * H_;

  char* ws = (char*)d_ws;
  size_t off = 0;
  auto carve = [&](size_t bytes) {
    char* p = ws + off;
    off += (bytes + 255) & ~(size_t)255;
    return p;
  };
  __bf16* hsb   = (__bf16*)carve(sizeof(__bf16) * (size_t)M_ * H_);
  __bf16* wb    = (__bf16*)carve(sizeof(__bf16) * 3 * (size_t)H_ * H_);
  __bf16* qb    = (__bf16*)carve(sizeof(__bf16) * (size_t)M_ * H_);
  __bf16* kb    = (__bf16*)carve(sizeof(__bf16) * (size_t)M_ * H_);
  __bf16* vb    = (__bf16*)carve(sizeof(__bf16) * (size_t)M_ * H_);
  __bf16* kselh = (__bf16*)carve(sizeof(__bf16) * (size_t)B_ * NH_ * KPH_ * DH_);
  __bf16* vTh   = (__bf16*)carve(sizeof(__bf16) * (size_t)B_ * NH_ * DH_ * KPH_);
  __bf16* ksell = (__bf16*)carve(sizeof(__bf16) * (size_t)B_ * NH_ * KPL_ * DH_);
  __bf16* vTl   = (__bf16*)carve(sizeof(__bf16) * (size_t)B_ * NH_ * DH_ * KPL_);
  float*  mh    = (float*)carve(sizeof(float) * (size_t)B_ * KPH_);
  float*  ml    = (float*)carve(sizeof(float) * (size_t)B_ * KPL_);
  float*  ctxh  = (float*)carve(sizeof(float) * (size_t)M_ * H_);
  float*  ctxl  = (float*)carve(sizeof(float) * (size_t)M_ * H_);

  // 1) bf16 conversions
  int nHS = M_ * H_;
  int nW  = H_ * H_;
  cvt_bf16_kernel<<<(nHS + 255) / 256, 256, 0, stream>>>(hs, hsb, nHS);
  cvt_bf16_kernel<<<(nW + 255) / 256, 256, 0, stream>>>(Wq, wb, nW);
  cvt_bf16_kernel<<<(nW + 255) / 256, 256, 0, stream>>>(Wk, wb + (size_t)nW, nW);
  cvt_bf16_kernel<<<(nW + 255) / 256, 256, 0, stream>>>(Wv, wb + 2 * (size_t)nW, nW);

  // 2) QKV projection
  int tiles = 3 * MT_ * NT_;
  qkv_gemm_kernel<<<(tiles + 7) / 8, 256, 0, stream>>>(hsb, wb, bq, bk, bv, qb, kb, vb);

  // 3) routing / gather
  route_kernel<<<B_ * NH_, 256, 0, stream>>>(qb, kb, vb, amask,
                                             kselh, vTh, ksell, vTl, mh, ml);

  // 4) high branch (writes attention_probs), 5) low branch
  int ablocks = B_ * NH_ * (S_ / 16);
  attn_kernel<KH_, KPH_, true><<<ablocks, 32, 16 * KPH_ * 6, stream>>>(
      qb, kselh, vTh, mh, out_probs, ctxh);
  attn_kernel<KL_, KPL_, false><<<ablocks, 32, 16 * KPL_ * 6, stream>>>(
      qb, ksell, vTl, ml, nullptr, ctxl);

  // 6) combine + transpose to (B,S,H)
  combine_kernel<<<(nHS + 255) / 256, 256, 0, stream>>>(ctxh, ctxl, out_ctx);
}
